// MergerSingleW_37649683317095
// MI455X (gfx1250) — matrix operand order repaired and checked by reading, not compiled, not run on hardware
//
#include <hip/hip_runtime.h>

// ---------------------------------------------------------------------------
// MergerSingleW fused kernel for gfx1250 (MI455X)
//   out = (x @ Wq + b1) @ Wq^T + b2,  Wq = alpha * Q, Q integer in [-63,63]\{0}
// Q stored exactly in f16; both GEMMs via v_wmma_f32_16x16x32_f16 with f32
// accumulation; full fusion so the 512MB 'h' intermediate never touches HBM.
// 32 rows per wave: 8 WMMAs per loop iteration against one set of B fragments.
// ---------------------------------------------------------------------------

typedef __attribute__((ext_vector_type(16))) _Float16 v16h;
typedef __attribute__((ext_vector_type(8)))  _Float16 v8h;
typedef __attribute__((ext_vector_type(8)))  float    v8f;

#define NROWS 65536
#define HDIM  2048
#define KDIM  32

__device__ __forceinline__ float softplus_alpha(float a) {
  // softplus(a) + 1e-6, numerically stable
  float sp = (a > 15.f) ? a : log1pf(expf(a));
  return sp + 1e-6f;
}

// Quantize W -> integer levels, scatter as f16 into WMMA B-fragment layouts.
// b1frag: GEMM1 B = Q [32 x H], tiled 32x16 (K x N) per 16 columns of H.
//   16-bit B 32x16 layout: lanes 0-15 hold K={0..7,16..23}, lanes 16-31
//   hold K={8..15,24..31}; lane's column = lane&15.
// b2frag: GEMM2 B = Q^T [H x 32], tiled per (K-step t of 32, col-tile n of 16).
__global__ void __launch_bounds__(256)
quantize_kernel(const float* __restrict__ W,
                const float* __restrict__ alpha_raw,
                _Float16* __restrict__ b1frag,
                _Float16* __restrict__ b2frag) {
  int idx = blockIdx.x * blockDim.x + threadIdx.x;   // 0 .. 32*2048-1
  int k = idx >> 11;        // row of W  (0..31)  = K index of GEMM1
  int c = idx & (HDIM - 1); // col of W  (0..2047)

  float alpha = softplus_alpha(alpha_raw[0]);
  float r = W[idx] / alpha;
  float n = rintf(r);
  n = fminf(fmaxf(n, -63.f), 63.f);
  if (n == 0.f) n = (r > 0.f) ? 1.f : -1.f;  // gap at 0; tie -> first index (-1)
  _Float16 q = (_Float16)n;

  // GEMM1 B fragment
  {
    int j    = c >> 4;                       // H tile
    int lane = (((k >> 3) & 1) << 4) | (c & 15);
    int e    = (k & 7) | ((k >> 4) << 3);
    b1frag[(j * 32 + lane) * 16 + e] = q;
  }
  // GEMM2 B fragment: B2[kk][col2] = Q[col2][kk] with kk=c, col2=k
  {
    int t    = c >> 5;                       // 32-wide K step along H
    int kl   = c & 31;
    int nt   = k >> 4;                       // which 16-col tile of the 32 outs
    int lane = (((kl >> 3) & 1) << 4) | (k & 15);
    int e    = (kl & 7) | ((kl >> 4) << 3);
    b2frag[((t * 2 + nt) * 32 + lane) * 16 + e] = q;
  }
}

#define WAVES_PER_BLOCK 8
#define ROWS_PER_WAVE   32
#define ROWS_PER_BLOCK  (WAVES_PER_BLOCK * ROWS_PER_WAVE)   // 256

__global__ void __launch_bounds__(256)
fused_gemm_kernel(const float* __restrict__ x,
                  const float* __restrict__ b1,
                  const float* __restrict__ b2,
                  const float* __restrict__ alpha_raw,
                  const _Float16* __restrict__ b1frag,
                  const _Float16* __restrict__ b2frag,
                  float* __restrict__ out) {
  // per-wave 32x32 f16 staging tile for the C-layout -> A-layout shuffle
  __shared__ _Float16 hstage[WAVES_PER_BLOCK][ROWS_PER_WAVE][32];

  const int lane = threadIdx.x & 31;
  const int wave = threadIdx.x >> 5;
  const int row0 = blockIdx.x * ROWS_PER_BLOCK + wave * ROWS_PER_WAVE;

  const float alpha = softplus_alpha(alpha_raw[0]);

  // ---- A fragments for GEMM1: x[row0 .. row0+31][0..31] as f16 ----
  // 16-bit A 16x32 layout: lane<16: row=lane,  K = {0..7, 16..23}
  //                        lane>=16: row=lane-16, K = {8..15, 24..31}
  const int mrow = lane & 15;
  const int koff = (lane >> 4) * 8;       // 0 or 8
  v16h afrag0, afrag1;
  {
    const float* xr0 = x + (row0 + mrow) * KDIM;
    const float* xr1 = x + (row0 + 16 + mrow) * KDIM;
#pragma unroll
    for (int i = 0; i < 8; ++i) {
      afrag0[i]     = (_Float16)xr0[koff + i];
      afrag0[8 + i] = (_Float16)xr0[16 + koff + i];
      afrag1[i]     = (_Float16)xr1[koff + i];
      afrag1[8 + i] = (_Float16)xr1[16 + koff + i];
    }
  }

  // f32 C/D 16x16 layout: lane l, VGPR v -> row = v + 8*(l>>4), col = l&15
  const int ccol     = lane & 15;
  const int crowbase = (lane >> 4) * 8;

  v8f acc00 = {};  // rows 0..15,  out cols 0..15
  v8f acc01 = {};  // rows 0..15,  out cols 16..31
  v8f acc10 = {};  // rows 16..31, out cols 0..15
  v8f acc11 = {};  // rows 16..31, out cols 16..31

  for (int jp = 0; jp < HDIM / 32; ++jp) {
#pragma unroll
    for (int jj = 0; jj < 2; ++jj) {
      const int j = jp * 2 + jj;
      // GEMM1: two 16x16 h tiles (row groups 0 and 1) share one B fragment
      v16h bfrag = *(const v16h*)(b1frag + (j * 32 + lane) * 16);
      v8f c0 = {}, c1 = {};
      c0 = __builtin_amdgcn_wmma_f32_16x16x32_f16(
          false, afrag0, false, bfrag, (short)0, c0, false, false);
      c1 = __builtin_amdgcn_wmma_f32_16x16x32_f16(
          false, afrag1, false, bfrag, (short)0, c1, false, false);
      // epilogue 1: h = alpha*acc + b1 ; stage as f16 (row-major) in LDS
      const float bb = b1[j * 16 + ccol];
#pragma unroll
      for (int v = 0; v < 8; ++v) {
        float h0 = alpha * c0[v] + bb;
        float h1 = alpha * c1[v] + bb;
        hstage[wave][crowbase + v][jj * 16 + ccol]      = (_Float16)h0;
        hstage[wave][16 + crowbase + v][jj * 16 + ccol] = (_Float16)h1;
      }
    }

    // same-wave LDS RAW: make the staged h tiles visible before A2 loads
    asm volatile("s_wait_dscnt 0" ::: "memory");

    // ---- A fragments for GEMM2 from LDS (K = jp*32 .. jp*32+31) ----
    v16h a2_0, a2_1;
    {
      const _Float16* base0 = &hstage[wave][mrow][0];
      const _Float16* base1 = &hstage[wave][16 + mrow][0];
      v8h lo0 = *(const v8h*)(base0 + koff);
      v8h hi0 = *(const v8h*)(base0 + 16 + koff);
      v8h lo1 = *(const v8h*)(base1 + koff);
      v8h hi1 = *(const v8h*)(base1 + 16 + koff);
      a2_0 = __builtin_shufflevector(lo0, hi0, 0, 1, 2, 3, 4, 5, 6, 7,
                                     8, 9, 10, 11, 12, 13, 14, 15);
      a2_1 = __builtin_shufflevector(lo1, hi1, 0, 1, 2, 3, 4, 5, 6, 7,
                                     8, 9, 10, 11, 12, 13, 14, 15);
    }

    // GEMM2: 4 accumulating WMMAs share one pair of B fragments
    v16h b2a = *(const v16h*)(b2frag + ((jp * 2 + 0) * 32 + lane) * 16);
    v16h b2b = *(const v16h*)(b2frag + ((jp * 2 + 1) * 32 + lane) * 16);
    acc00 = __builtin_amdgcn_wmma_f32_16x16x32_f16(
        false, a2_0, false, b2a, (short)0, acc00, false, false);
    acc10 = __builtin_amdgcn_wmma_f32_16x16x32_f16(
        false, a2_1, false, b2a, (short)0, acc10, false, false);
    acc01 = __builtin_amdgcn_wmma_f32_16x16x32_f16(
        false, a2_0, false, b2b, (short)0, acc01, false, false);
    acc11 = __builtin_amdgcn_wmma_f32_16x16x32_f16(
        false, a2_1, false, b2b, (short)0, acc11, false, false);
  }

  // ---- epilogue 2: out = alpha*acc + b2 ----
  const float bb2a = b2[ccol];
  const float bb2b = b2[16 + ccol];
#pragma unroll
  for (int v = 0; v < 8; ++v) {
    const int r0 = row0 + crowbase + v;
    const int r1 = row0 + 16 + crowbase + v;
    out[r0 * 32 + ccol]      = alpha * acc00[v] + bb2a;
    out[r0 * 32 + 16 + ccol] = alpha * acc01[v] + bb2b;
    out[r1 * 32 + ccol]      = alpha * acc10[v] + bb2a;
    out[r1 * 32 + 16 + ccol] = alpha * acc11[v] + bb2b;
  }
}

extern "C" void kernel_launch(void* const* d_in, const int* in_sizes, int n_in,
                              void* d_out, int out_size, void* d_ws, size_t ws_size,
                              hipStream_t stream) {
  const float* x         = (const float*)d_in[0];
  const float* W         = (const float*)d_in[1];
  const float* b1        = (const float*)d_in[2];
  const float* b2        = (const float*)d_in[3];
  const float* alpha_raw = (const float*)d_in[4];
  float* out = (float*)d_out;

  _Float16* b1frag = (_Float16*)d_ws;                 // 32*2048 f16 = 128 KB
  _Float16* b2frag = b1frag + (size_t)KDIM * HDIM;    // another 128 KB

  quantize_kernel<<<(KDIM * HDIM) / 256, 256, 0, stream>>>(W, alpha_raw,
                                                           b1frag, b2frag);
  fused_gemm_kernel<<<NROWS / ROWS_PER_BLOCK, 256, 0, stream>>>(
      x, b1, b2, alpha_raw, b1frag, b2frag, out);
}
